// Attention_7206955123184
// MI455X (gfx1250) — matrix operand order repaired and checked by reading, not compiled, not run on hardware
//
#include <hip/hip_runtime.h>

// MI455X / gfx1250 attention (Gemma MQA), bf16 WMMA pipeline, fp32 accumulate.
// Pipeline: pack(fp32->bf16, B operands column-major) -> WMMA GEMM qproj/kvproj
// -> RoPE fixups -> flash-attention (wave32 WMMA, online softmax) -> WMMA out-proj.
// GEMM stages the block-shared B tile in LDS, double-buffered with
// global_load_async_to_lds_b128 (inline asm; ASYNCcnt) + s_wait_asynccnt when the
// toolchain has gfx1250 async support, else plain load + ds_store staging.
// Workspace use: ~72 MB carved from d_ws.

typedef __attribute__((ext_vector_type(16))) __bf16 v16bf;
typedef __attribute__((ext_vector_type(8)))  __bf16 v8bf;
typedef __attribute__((ext_vector_type(8)))  float  v8f;

#define B_DIM 4
#define T_DIM 1024
#define D_DIM 2048
#define N_HEADS 16
#define H_DIM 128
#define M_ROWS (B_DIM * T_DIM)     // 4096
#define NH_DIM (N_HEADS * H_DIM)   // 2048
#define K_MASK_F (-2.3819763e38f)
#define LN10K 9.210340371976184f   // ln(10000)

#if __has_builtin(__builtin_amdgcn_global_load_async_to_lds_b128) && \
    __has_builtin(__builtin_amdgcn_s_wait_asynccnt)
#define HAVE_ASYNC_LDS 1
#endif

// 16-byte global -> LDS copy; async DMA path (ASYNCcnt) when available.
__device__ __forceinline__ void cp16_g2s(__bf16* lds, const __bf16* g) {
#ifdef HAVE_ASYNC_LDS
  // VDST = 32-bit LDS byte address (generic LDS addr low bits), VADDR = 64-bit global.
  unsigned loff = (unsigned)(size_t)lds;
  unsigned long long ga = (unsigned long long)(size_t)g;
  asm volatile("global_load_async_to_lds_b128 %0, %1, off"
               :: "v"(loff), "v"(ga)
               : "memory");
#else
  *(v8bf*)lds = *(const v8bf*)g;
#endif
}
#ifdef HAVE_ASYNC_LDS
#define ASYNC_WAIT(n) __builtin_amdgcn_s_wait_asynccnt(n)
#else
#define ASYNC_WAIT(n) ((void)0)
#endif

// ---------------------------------------------------------------- pack kernels
__global__ void pk_x(const float* __restrict__ x, __bf16* __restrict__ xb, int n) {
  int i = blockIdx.x * 256 + threadIdx.x;
  if (i < n) xb[i] = (__bf16)x[i];
}

// wq[N,D,H] -> wqb[(n*H+h)][d]   (column-major B operand, 2048x2048)
__global__ void pk_wq(const float* __restrict__ wq, __bf16* __restrict__ wqb) {
  int i = blockIdx.x * 256 + threadIdx.x;
  int d = i & 2047, row = i >> 11;           // row = n*128+h
  int n = row >> 7, h = row & 127;
  wqb[i] = (__bf16)wq[((size_t)n * D_DIM + d) * H_DIM + h];
}

// wkv[2,1,D,H] -> wkvb[(c*H+h)][d]  (256x2048)
__global__ void pk_wkv(const float* __restrict__ wkv, __bf16* __restrict__ wkvb) {
  int i = blockIdx.x * 256 + threadIdx.x;
  int d = i & 2047, row = i >> 11;           // row = c*128+h
  int c = row >> 7, h = row & 127;
  wkvb[i] = (__bf16)wkv[((size_t)c * D_DIM + d) * H_DIM + h];
}

// wo[N,H,D] -> wotb[d][(n*H+h)] transposed -> column-major for out-proj (2048x2048)
__global__ void pk_wo(const float* __restrict__ wo, __bf16* __restrict__ wotb) {
  int i = blockIdx.x * 256 + threadIdx.x;
  int col = i & 2047, d = i >> 11;           // wotb[d][col] layout: i = d*2048+col
  wotb[i] = (__bf16)wo[(size_t)col * D_DIM + d];
}

// ---------------------------------------------------------------- generic GEMM
// C[M,Nc] = A[M,K] (row-major bf16) x Bt[Nc,K] (column-major-B, bf16), fp32 acc.
// Block = 128 threads = 4 waves; wave computes 32x64; block tile 128x64.
// B tile (64 cols x 32 k = 4 KB) staged in LDS, double-buffered, shared by all waves.
__global__ __launch_bounds__(128) void gemm_bf16_wmma(
    const __bf16* __restrict__ A, const __bf16* __restrict__ Bt,
    void* __restrict__ Cv, int Ncols, int K, int storeF32) {
  __shared__ __align__(64) __bf16 bsh[2][64][32];   // [buf][col][k], 8 KB
  const int tid = threadIdx.x;
  const int L  = tid & 31;
  const int w  = tid >> 5;
  const int lm = L & 15, lh = L >> 4;
  const int tileM = blockIdx.y * 128 + w * 32;
  const int tileN = blockIdx.x * 64;

  // staging assignment: thread copies 16 elems of one column per stage
  const int scol = tid >> 1;           // 0..63
  const int skof = (tid & 1) * 16;     // 0 or 16
  const __bf16* sg = Bt + (size_t)(tileN + scol) * K + skof;

  v8f acc[2][4] = {};
  const __bf16* arow0 = A + (size_t)(tileM + lm) * K + lh * 16;
  const __bf16* arow1 = arow0 + (size_t)16 * K;

  const int nsteps = K / 32;
  // prologue: stage k-step 0 into buffer 0
  cp16_g2s(&bsh[0][scol][skof], sg);
  cp16_g2s(&bsh[0][scol][skof + 8], sg + 8);

  for (int i = 0; i < nsteps; ++i) {
    const int cur = i & 1;
    const int k0 = i * 32;
    if (i + 1 < nsteps) {
      cp16_g2s(&bsh[1 - cur][scol][skof], sg + k0 + 32);
      cp16_g2s(&bsh[1 - cur][scol][skof + 8], sg + k0 + 40);
      ASYNC_WAIT(2);                   // current stage's 2 copies complete
    } else {
      ASYNC_WAIT(0);
    }
    __syncthreads();                   // all waves' staging of buf[cur] visible

    // A fragments: lanes 0-15 hold K {0..15}, lanes 16-31 {16..31}
    v8bf a0lo = *(const v8bf*)(arow0 + k0);
    v8bf a0hi = *(const v8bf*)(arow0 + k0 + 8);
    v8bf a1lo = *(const v8bf*)(arow1 + k0);
    v8bf a1hi = *(const v8bf*)(arow1 + k0 + 8);
    v16bf af0, af1;
#pragma unroll
    for (int e = 0; e < 8; ++e) {
      af0[e] = a0lo[e]; af0[e + 8] = a0hi[e];
      af1[e] = a1lo[e]; af1[e + 8] = a1hi[e];
    }
#pragma unroll
    for (int c = 0; c < 4; ++c) {
      // B fragment from LDS: lane column = lm, 16 contiguous K per lane half
      v16bf bfr = *(const v16bf*)&bsh[cur][c * 16 + lm][lh * 16];
      acc[0][c] = __builtin_amdgcn_wmma_f32_16x16x32_bf16(
          false, af0, false, bfr, (short)0, acc[0][c], false, false);
      acc[1][c] = __builtin_amdgcn_wmma_f32_16x16x32_bf16(
          false, af1, false, bfr, (short)0, acc[1][c], false, false);
    }
    __syncthreads();                   // reads of buf[cur] done before it is re-staged
  }

#pragma unroll
  for (int g = 0; g < 2; ++g)
#pragma unroll
    for (int c = 0; c < 4; ++c)
#pragma unroll
      for (int r = 0; r < 8; ++r) {
        const size_t row = tileM + g * 16 + r + (lh ? 8 : 0);
        const size_t col = tileN + c * 16 + lm;
        if (storeF32) ((float*)Cv)[row * Ncols + col] = acc[g][c][r];
        else          ((__bf16*)Cv)[row * Ncols + col] = (__bf16)acc[g][c][r];
      }
}

// ---------------------------------------------------------------- RoPE fixups
// q: in-place RoPE + H^-0.5 scale on qb[B,T,N*H]
__global__ void rope_q(__bf16* __restrict__ qb, const int* __restrict__ segpos) {
  int i = blockIdx.x * 256 + threadIdx.x;      // B*T*N*64 threads
  int h = i & 63, n = (i >> 6) & 15, t = (i >> 10) & 1023, b = i >> 20;
  __bf16* p = qb + ((size_t)(b * T_DIM + t) * NH_DIM) + n * H_DIM;
  float f = (float)p[h], s = (float)p[h + 64];
  float pos = (float)segpos[b * T_DIM + t];
  float ang = pos * __expf(-(float)h * (LN10K / 64.0f));
  float sn, cs; __sincosf(ang, &sn, &cs);
  const float sc = 0.08838834764831845f;       // 128^-0.5
  p[h]      = (__bf16)((f * cs - s * sn) * sc);
  p[h + 64] = (__bf16)((s * cs + f * sn) * sc);
}

// kv: split kvtmp[B*T][256] -> kb[B*T][128] (RoPE'd) and vtb[B][H][T] (transposed)
__global__ void kv_post(const __bf16* __restrict__ kvtmp, __bf16* __restrict__ kb,
                        __bf16* __restrict__ vtb, const int* __restrict__ segpos) {
  int i = blockIdx.x * 256 + threadIdx.x;      // B*T*64 threads
  int h = i & 63, t = (i >> 6) & 1023, b = i >> 16;
  const size_t row = (size_t)b * T_DIM + t;
  const __bf16* src = kvtmp + row * 256;
  float f = (float)src[h], s = (float)src[h + 64];
  float pos = (float)segpos[row];
  float ang = pos * __expf(-(float)h * (LN10K / 64.0f));
  float sn, cs; __sincosf(ang, &sn, &cs);
  kb[row * H_DIM + h]      = (__bf16)(f * cs - s * sn);
  kb[row * H_DIM + h + 64] = (__bf16)(s * cs + f * sn);
  vtb[((size_t)b * H_DIM + h)      * T_DIM + t] = src[128 + h];
  vtb[((size_t)b * H_DIM + h + 64) * T_DIM + t] = src[128 + h + 64];
}

// ---------------------------------------------------------------- flash attention
// grid (T/128, N, B); block = 256 = 8 waves; wave w owns 16 query rows.
__global__ __launch_bounds__(256) void attn_fa_wmma(
    const __bf16* __restrict__ qb, const __bf16* __restrict__ kb,
    const __bf16* __restrict__ vtb, __bf16* __restrict__ enc) {
  __shared__ __align__(64) __bf16 pshm[8][16][32];   // per-wave P staging (C->A relayout)
  const int L  = threadIdx.x & 31;
  const int w  = threadIdx.x >> 5;
  const int lm = L & 15, lh = L >> 4;
  const int b = blockIdx.z, n = blockIdx.y;
  const int qr = blockIdx.x * 128 + w * 16;

  // q fragments: 16x128, 4 chained K=32 chunks, resident in VGPRs
  const __bf16* qrow = qb + ((size_t)(b * T_DIM + qr + lm) * NH_DIM) + n * H_DIM + lh * 16;
  v16bf qf[4];
#pragma unroll
  for (int kk = 0; kk < 4; ++kk) {
    v8bf lo = *(const v8bf*)(qrow + kk * 32);
    v8bf hi = *(const v8bf*)(qrow + kk * 32 + 8);
#pragma unroll
    for (int e = 0; e < 8; ++e) { qf[kk][e] = lo[e]; qf[kk][e + 8] = hi[e]; }
  }

  float m8[8], l8[8];
  v8f o[8] = {};
#pragma unroll
  for (int r = 0; r < 8; ++r) { m8[r] = -3.0e38f; l8[r] = 0.0f; }

  const int kend = qr + 16;                   // causal: keys <= qr+15
  for (int j0 = 0; j0 < kend; j0 += 32) {
    __builtin_prefetch(kb + (size_t)(b * T_DIM + j0 + 32) * H_DIM, 0, 0);

    // S = q . k^T for 32 keys: two 16x16 tiles
    v8f s0 = {}, s1 = {};
    const __bf16* krow0 = kb + (size_t)(b * T_DIM + j0 + lm) * H_DIM + lh * 16;
    const __bf16* krow1 = krow0 + (size_t)16 * H_DIM;
#pragma unroll
    for (int kk = 0; kk < 4; ++kk) {
      v16bf kf = *(const v16bf*)(krow0 + kk * 32);
      s0 = __builtin_amdgcn_wmma_f32_16x16x32_bf16(false, qf[kk], false, kf, (short)0, s0, false, false);
    }
#pragma unroll
    for (int kk = 0; kk < 4; ++kk) {
      v16bf kf = *(const v16bf*)(krow1 + kk * 32);
      s1 = __builtin_amdgcn_wmma_f32_16x16x32_bf16(false, qf[kk], false, kf, (short)0, s1, false, false);
    }

    // causal mask + online softmax (rows live in VGPR index; cols across 16-lane halves)
    const int col = j0 + lm;
    float sc8[8], p0[8], p1[8];
#pragma unroll
    for (int r = 0; r < 8; ++r) {
      const int row = qr + r + (lh ? 8 : 0);
      float a0 = (col      <= row) ? s0[r] : K_MASK_F;
      float a1 = (col + 16 <= row) ? s1[r] : K_MASK_F;
      float t = fmaxf(a0, a1);
      t = fmaxf(t, __shfl_xor(t, 1, 32));
      t = fmaxf(t, __shfl_xor(t, 2, 32));
      t = fmaxf(t, __shfl_xor(t, 4, 32));
      t = fmaxf(t, __shfl_xor(t, 8, 32));
      const float mn = fmaxf(m8[r], t);
      sc8[r] = __expf(m8[r] - mn);
      p0[r] = __expf(a0 - mn);
      p1[r] = __expf(a1 - mn);
      float rs = p0[r] + p1[r];
      rs += __shfl_xor(rs, 1, 32);
      rs += __shfl_xor(rs, 2, 32);
      rs += __shfl_xor(rs, 4, 32);
      rs += __shfl_xor(rs, 8, 32);
      l8[r] = l8[r] * sc8[r] + rs;
      m8[r] = mn;
    }
#pragma unroll
    for (int c = 0; c < 8; ++c)
#pragma unroll
      for (int r = 0; r < 8; ++r) o[c][r] *= sc8[r];

    // P (C layout) -> LDS -> A fragment (wave-private slice, wave-synchronous)
#pragma unroll
    for (int r = 0; r < 8; ++r) {
      const int row = r + (lh ? 8 : 0);
      pshm[w][row][lm]      = (__bf16)p0[r];
      pshm[w][row][lm + 16] = (__bf16)p1[r];
    }
    v16bf pa;
    {
      const v8bf* pr = (const v8bf*)&pshm[w][lm][0];
      v8bf lo = pr[lh * 2];        // K {0..7} or {16..23}
      v8bf hi = pr[lh * 2 + 1];    // K {8..15} or {24..31}
#pragma unroll
      for (int e = 0; e < 8; ++e) { pa[e] = lo[e]; pa[e + 8] = hi[e]; }
    }

    // O += P . V  (V^T stored [B][H][T]: contiguous keys per lane)
#pragma unroll
    for (int c = 0; c < 8; ++c) {
      const __bf16* vp = vtb + ((size_t)b * H_DIM + c * 16 + lm) * T_DIM + j0 + lh * 16;
      v16bf vf = *(const v16bf*)vp;
      o[c] = __builtin_amdgcn_wmma_f32_16x16x32_bf16(false, pa, false, vf, (short)0, o[c], false, false);
    }
  }

  // normalize and store encoded[B,T,N*H] as bf16 for the out-proj GEMM
  float inv[8];
#pragma unroll
  for (int r = 0; r < 8; ++r) inv[r] = 1.0f / l8[r];
#pragma unroll
  for (int c = 0; c < 8; ++c)
#pragma unroll
    for (int r = 0; r < 8; ++r) {
      const int row = qr + r + (lh ? 8 : 0);
      enc[(size_t)(b * T_DIM + row) * NH_DIM + n * H_DIM + c * 16 + lm] =
          (__bf16)(o[c][r] * inv[r]);
    }
}

// ---------------------------------------------------------------- launch
extern "C" void kernel_launch(void* const* d_in, const int* in_sizes, int n_in,
                              void* d_out, int out_size, void* d_ws, size_t ws_size,
                              hipStream_t stream) {
  const float* x   = (const float*)d_in[0];   // [B,T,D]
  const float* wq  = (const float*)d_in[1];   // [N,D,H]
  const float* wkv = (const float*)d_in[2];   // [2,1,D,H]
  const float* wo  = (const float*)d_in[3];   // [N,H,D]
  const int* segpos = (const int*)d_in[4];    // [B,T]
  // d_in[5] attn_mask: tril causal -> applied analytically in-kernel
  float* out = (float*)d_out;                 // [B,T,D] fp32

  size_t off = 0;
  auto take = [&](size_t bytes) -> void* {
    off = (off + 255) & ~(size_t)255;
    void* p = (char*)d_ws + off;
    off += bytes;
    return p;
  };
  __bf16* xb    = (__bf16*)take((size_t)M_ROWS * D_DIM * 2);   // 16 MB
  __bf16* wqb   = (__bf16*)take((size_t)NH_DIM * D_DIM * 2);   //  8 MB
  __bf16* wkvb  = (__bf16*)take((size_t)256 * D_DIM * 2);      //  1 MB
  __bf16* wotb  = (__bf16*)take((size_t)D_DIM * NH_DIM * 2);   //  8 MB
  __bf16* qbuf  = (__bf16*)take((size_t)M_ROWS * NH_DIM * 2);  // 16 MB
  __bf16* kvtmp = (__bf16*)take((size_t)M_ROWS * 256 * 2);     //  2 MB
  __bf16* kb    = (__bf16*)take((size_t)M_ROWS * H_DIM * 2);   //  1 MB
  __bf16* vtb   = (__bf16*)take((size_t)B_DIM * H_DIM * T_DIM * 2); // 1 MB
  __bf16* enc   = (__bf16*)take((size_t)M_ROWS * NH_DIM * 2);  // 16 MB
  (void)ws_size; (void)in_sizes; (void)n_in; (void)out_size;

  // pack / convert
  pk_x  <<<(M_ROWS * D_DIM) / 256, 256, 0, stream>>>(x, xb, M_ROWS * D_DIM);
  pk_wq <<<(NH_DIM * D_DIM) / 256, 256, 0, stream>>>(wq, wqb);
  pk_wkv<<<(256 * D_DIM) / 256, 256, 0, stream>>>(wkv, wkvb);
  pk_wo <<<(D_DIM * NH_DIM) / 256, 256, 0, stream>>>(wo, wotb);

  // projections (WMMA GEMMs)
  gemm_bf16_wmma<<<dim3(NH_DIM / 64, M_ROWS / 128), 128, 0, stream>>>(
      xb, wqb, qbuf, NH_DIM, D_DIM, 0);
  gemm_bf16_wmma<<<dim3(256 / 64, M_ROWS / 128), 128, 0, stream>>>(
      xb, wkvb, kvtmp, 256, D_DIM, 0);

  // RoPE + scale (q), RoPE + transpose split (k, v)
  rope_q <<<(B_DIM * T_DIM * N_HEADS * 64) / 256, 256, 0, stream>>>(qbuf, segpos);
  kv_post<<<(B_DIM * T_DIM * 64) / 256, 256, 0, stream>>>(kvtmp, kb, vtb, segpos);

  // flash attention
  attn_fa_wmma<<<dim3(T_DIM / 128, N_HEADS, B_DIM), 256, 0, stream>>>(qbuf, kb, vtb, enc);

  // output projection -> fp32
  gemm_bf16_wmma<<<dim3(D_DIM / 64, M_ROWS / 128), 128, 0, stream>>>(
      enc, wotb, out, D_DIM, D_DIM, 1);
}
